// ModernFinGATWithSector_29686813950823
// MI455X (gfx1250) — matrix operand order repaired and checked by reading, not compiled, not run on hardware
//
#include <hip/hip_runtime.h>
#include <hip/hip_bf16.h>

typedef __attribute__((ext_vector_type(16))) _Float16 v16h;
typedef __attribute__((ext_vector_type(8)))  _Float16 v8h;
typedef __attribute__((ext_vector_type(8)))  float    v8f;
typedef __attribute__((ext_vector_type(4)))  float    f32x4;

#define HID 256
#define ENC_NEG_INF 0x007FFFFFu

__device__ __forceinline__ unsigned fenc(float f) {
  unsigned u = __float_as_uint(f);
  return (u & 0x80000000u) ? ~u : (u | 0x80000000u);
}
__device__ __forceinline__ float fdec(unsigned e) {
  unsigned u = (e & 0x80000000u) ? (e & 0x7fffffffu) : ~e;
  return __uint_as_float(u);
}
__device__ __forceinline__ float wave_sum(float v) {
#pragma unroll
  for (int o = 16; o > 0; o >>= 1) v += __shfl_xor(v, o, 32);
  return v;
}

// ---------------------------------------------------------------- fills
__global__ void fill_f32_kernel(float* p, float v, size_t n) {
  size_t i = (size_t)blockIdx.x * 256 + threadIdx.x;
  if (i < n) p[i] = v;
}
__global__ void fill_u32_kernel(unsigned* p, unsigned v, size_t n) {
  size_t i = (size_t)blockIdx.x * 256 + threadIdx.x;
  if (i < n) p[i] = v;
}

// -------------------------------------------- weight convert + transpose
// W: f32 [K][Nout] row-major  ->  O: f16 [Nout][K] row-major
__global__ void cvt_transpose_kernel(const float* __restrict__ W,
                                     _Float16* __restrict__ O, int K, int Nout) {
  int i = blockIdx.x * 256 + threadIdx.x;
  if (i >= K * Nout) return;
  int n = i / K, k = i - n * K;
  O[i] = (_Float16)W[(size_t)k * Nout + n];
}

// ---------------------------------------------------------- edge builder
__global__ void build_edges_kernel(const int* __restrict__ ei, int* __restrict__ src,
                                   int* __restrict__ dst, int E, int Nn) {
  int i = blockIdx.x * 256 + threadIdx.x;
  int tot = E + Nn;
  if (i >= tot) return;
  if (i < E) { src[i] = ei[i]; dst[i] = ei[E + i]; }
  else       { src[i] = i - E; dst[i] = i - E; }
}

// --------------------------------------------------- layernorm + act
// act: 0 none, 1 elu, 2 relu.  t = X + res + bvec ; Y = act(ln(t)*g+b)
__global__ __launch_bounds__(128) void ln_act_kernel(
    const float* __restrict__ X, const float* __restrict__ res,
    const float* __restrict__ bvec, const float* __restrict__ g,
    const float* __restrict__ b, float* __restrict__ Y, int M, int D, int act) {
  int row  = blockIdx.x * 4 + (threadIdx.x >> 5);
  int lane = threadIdx.x & 31;
  if (row >= M) return;
  const float* xp = X + (size_t)row * D;
  const float* rp = res ? res + (size_t)row * D : nullptr;
  float vals[8];
  int per = D >> 5;
  float s = 0.f;
  for (int j = 0; j < per; ++j) {
    int d = lane + (j << 5);
    float t = xp[d];
    if (rp)   t += rp[d];
    if (bvec) t += bvec[d];
    vals[j] = t; s += t;
  }
  float mean = wave_sum(s) / (float)D;
  float vs = 0.f;
  for (int j = 0; j < per; ++j) { float dd = vals[j] - mean; vs += dd * dd; }
  float rstd = rsqrtf(wave_sum(vs) / (float)D + 1e-5f);
  for (int j = 0; j < per; ++j) {
    int d = lane + (j << 5);
    float y = (vals[j] - mean) * rstd * g[d] + b[d];
    if (act == 1)      y = y > 0.f ? y : (__expf(y) - 1.f);
    else if (act == 2) y = fmaxf(y, 0.f);
    Y[(size_t)row * D + d] = y;
  }
}

// --------------------------------------------------------- WMMA GEMM
// C[M][Nout] = act(A[M][K] * Bt^T + bias), Bt: f16 [Nout][K] row-major.
// block = 128 threads (4 waves); block tile 128x64; wave tile 2x(16x64).
// B panel (64 cols x 32 k, 4KB f16) is staged in LDS per K-step via
// gfx1250 async global->LDS loads (ASYNCcnt), then fragments read from LDS.
template <int ACT>
__global__ __launch_bounds__(128) void wmma_gemm_kernel(
    const float* __restrict__ A, const _Float16* __restrict__ Bt,
    const float* __restrict__ bias, float* __restrict__ C,
    int M, int K, int Nout) {
  __shared__ _Float16 bsh[64 * 32];   // [col][k] panel, 4 KB

  const int tid  = threadIdx.x;
  const int lane = tid & 31;
  const int wv   = tid >> 5;
  const int rowBase0 = blockIdx.x * 128 + wv * 16;
  const int rowBase1 = rowBase0 + 64;
  const int colBase  = blockIdx.y * 64;
  const int kh = lane >> 4;   // lane half selects K sub-group
  const int lm = lane & 15;
  int arow0 = rowBase0 + lm; if (arow0 >= M) arow0 = M - 1;
  int arow1 = rowBase1 + lm; if (arow1 >= M) arow1 = M - 1;

  const unsigned ldsBase = (unsigned)(size_t)(void*)bsh;
  // this thread's two 16B chunks of the B panel (256 chunks total)
  const int ch0 = tid, ch1 = tid + 128;

  v8f acc[2][4];
#pragma unroll
  for (int s = 0; s < 2; ++s)
#pragma unroll
    for (int t = 0; t < 4; ++t) acc[s][t] = (v8f)(0.f);

  for (int k0 = 0; k0 < K; k0 += 32) {
    __syncthreads();   // previous iteration's LDS reads complete
    {
      // async copy B panel chunk ch: col = ch>>2, 16B piece j = ch&3
      int c0 = ch0 >> 2, j0 = ch0 & 3;
      int c1 = ch1 >> 2, j1 = ch1 & 3;
      unsigned long long g0 = (unsigned long long)(const void*)(Bt) +
          (((size_t)(colBase + c0) * K + k0) * 2 + (size_t)j0 * 16);
      unsigned long long g1 = (unsigned long long)(const void*)(Bt) +
          (((size_t)(colBase + c1) * K + k0) * 2 + (size_t)j1 * 16);
      unsigned l0 = ldsBase + (unsigned)ch0 * 16;
      unsigned l1 = ldsBase + (unsigned)ch1 * 16;
      asm volatile("global_load_async_to_lds_b128 %0, %1, off"
                   :: "v"(l0), "v"(g0) : "memory");
      asm volatile("global_load_async_to_lds_b128 %0, %1, off"
                   :: "v"(l1), "v"(g1) : "memory");
      asm volatile("s_wait_asynccnt 0" ::: "memory");
    }
    __syncthreads();   // panel fully resident for all waves

    // A fragments (f32 -> f16): 16-bit A 16x32 layout
    const float* pa0 = A + (size_t)arow0 * K + k0 + 16 * kh;
    const float* pa1 = A + (size_t)arow1 * K + k0 + 16 * kh;
    if (k0 + 32 < K) {
      __builtin_prefetch(pa0 + 32, 0, 1);
      __builtin_prefetch(pa1 + 32, 0, 1);
    }
    f32x4 a00 = *(const f32x4*)(pa0);
    f32x4 a01 = *(const f32x4*)(pa0 + 4);
    f32x4 a02 = *(const f32x4*)(pa0 + 8);
    f32x4 a03 = *(const f32x4*)(pa0 + 12);
    f32x4 a10 = *(const f32x4*)(pa1);
    f32x4 a11 = *(const f32x4*)(pa1 + 4);
    f32x4 a12 = *(const f32x4*)(pa1 + 8);
    f32x4 a13 = *(const f32x4*)(pa1 + 12);
    v16h af0, af1;
#pragma unroll
    for (int j = 0; j < 4; ++j) {
      af0[j]      = (_Float16)a00[j];
      af0[4 + j]  = (_Float16)a01[j];
      af0[8 + j]  = (_Float16)a02[j];
      af0[12 + j] = (_Float16)a03[j];
      af1[j]      = (_Float16)a10[j];
      af1[4 + j]  = (_Float16)a11[j];
      af1[8 + j]  = (_Float16)a12[j];
      af1[12 + j] = (_Float16)a13[j];
    }
#pragma unroll
    for (int t = 0; t < 4; ++t) {
      // B fragment: lanes 0-15 hold K=k0..k0+15 of column, lanes 16-31: +16
      int cl = t * 16 + lm;
      v16h bf = *(const v16h*)&bsh[cl * 32 + 16 * kh];
      acc[0][t] = __builtin_amdgcn_wmma_f32_16x16x32_f16(
          false, af0, false, bf, (short)0, acc[0][t], false, false);
      acc[1][t] = __builtin_amdgcn_wmma_f32_16x16x32_f16(
          false, af1, false, bf, (short)0, acc[1][t], false, false);
    }
  }

#pragma unroll
  for (int s = 0; s < 2; ++s) {
    int rb = s ? rowBase1 : rowBase0;
#pragma unroll
    for (int t = 0; t < 4; ++t) {
      int col = colBase + t * 16 + lm;
      float bb = bias ? bias[col] : 0.f;
#pragma unroll
      for (int r = 0; r < 8; ++r) {
        int row = rb + r + 8 * kh;
        if (row < M) {
          float v = acc[s][t][r] + bb;
          if (ACT == 1)      v = fmaxf(v, 0.f);
          else if (ACT == 2) v = tanhf(v);
          C[(size_t)row * Nout + col] = v;
        }
      }
    }
  }
}

// ------------------------------------------------ GATv2 edge kernels
__global__ __launch_bounds__(128) void edge_phase1_kernel(
    const float* __restrict__ xl, const float* __restrict__ xr,
    const float* __restrict__ att, const int* __restrict__ src,
    const int* __restrict__ dst, float* __restrict__ logits,
    unsigned* __restrict__ mxe, int Etot) {
  int e    = blockIdx.x * 4 + (threadIdx.x >> 5);
  int lane = threadIdx.x & 31;
  if (e >= Etot) return;
  int s = src[e], d = dst[e];
  const float* ps = xl + (size_t)s * HID;
  const float* pd = xr + (size_t)d * HID;
#pragma unroll
  for (int h = 0; h < 8; ++h) {
    float v = ps[h * 32 + lane] + pd[h * 32 + lane];
    v = v > 0.f ? v : 0.2f * v;                   // leaky_relu(0.2)
    float p = wave_sum(v * att[h * 32 + lane]);
    if (lane == 0) {
      logits[(size_t)e * 8 + h] = p;
      atomicMax(mxe + (size_t)d * 8 + h, fenc(p));
    }
  }
}
__global__ void edge_phase2_kernel(float* __restrict__ logits,
                                   const unsigned* __restrict__ mxe,
                                   float* __restrict__ den,
                                   const int* __restrict__ dst, int Etot) {
  int idx = blockIdx.x * 256 + threadIdx.x;
  if (idx >= Etot * 8) return;
  int e = idx >> 3, h = idx & 7;
  int d = dst[e];
  float ex = __expf(logits[idx] - fdec(mxe[(size_t)d * 8 + h]));
  logits[idx] = ex;
  atomicAdd(den + (size_t)d * 8 + h, ex);
}
__global__ __launch_bounds__(128) void edge_phase3_kernel(
    const float* __restrict__ xl, const float* __restrict__ ex,
    const float* __restrict__ den, const int* __restrict__ src,
    const int* __restrict__ dst, float* __restrict__ acc, int Etot) {
  int e    = blockIdx.x * 4 + (threadIdx.x >> 5);
  int lane = threadIdx.x & 31;
  if (e >= Etot) return;
  int s = src[e], d = dst[e];
  const float* ps = xl + (size_t)s * HID;
  float* pa = acc + (size_t)d * HID;
#pragma unroll
  for (int h = 0; h < 8; ++h) {
    float alpha = ex[(size_t)e * 8 + h] / den[(size_t)d * 8 + h];
    atomicAdd(pa + h * 32 + lane, ps[h * 32 + lane] * alpha);
  }
}

// ------------------------------------------------ small dense heads
__global__ __launch_bounds__(128) void small_head_kernel(
    const float* __restrict__ X, const float* __restrict__ W,
    const float* __restrict__ bias, float* __restrict__ out,
    int M, int D, int Nout, int ostride, int mode) {
  int row  = blockIdx.x * 4 + (threadIdx.x >> 5);
  int lane = threadIdx.x & 31;
  if (row >= M) return;
  const float* xp = X + (size_t)row * D;
  for (int n = 0; n < Nout; ++n) {
    float s = 0.f;
    for (int d = lane; d < D; d += 32) s += xp[d] * W[(size_t)d * Nout + n];
    s = wave_sum(s);
    if (lane == 0) {
      float v = s + (bias ? bias[n] : 0.f);
      if (mode == 1)      v = tanhf(v) * 0.1f;
      else if (mode == 2) v = 1.f / (1.f + __expf(-v));
      out[(size_t)row * ostride + n] = v;
    }
  }
}

// ------------------------------------------------ attention pooling
__global__ void seg_max_kernel(const float* __restrict__ s, const int* __restrict__ seg,
                               unsigned* __restrict__ mx, int n) {
  int i = blockIdx.x * 256 + threadIdx.x;
  if (i < n) atomicMax(mx + seg[i], fenc(s[i]));
}
__global__ void seg_exp_kernel(const float* __restrict__ s, const int* __restrict__ seg,
                               const unsigned* __restrict__ mx, float* __restrict__ es,
                               float* __restrict__ den, float* __restrict__ cnt, int n) {
  int i = blockIdx.x * 256 + threadIdx.x;
  if (i >= n) return;
  int b = seg[i];
  float e = __expf(s[i] - fdec(mx[b]));
  es[i] = e;
  atomicAdd(den + b, e);
  atomicAdd(cnt + b, 1.f);
}
__global__ void pool_acc_kernel(const float* __restrict__ x, const float* __restrict__ es,
                                const float* __restrict__ den, const int* __restrict__ seg,
                                float* __restrict__ sec, int n) {
  int i = blockIdx.x * 256 + threadIdx.x;
  if (i >= n) return;
  int r = i >> 8, d = i & 255;
  int b = seg[r];
  atomicAdd(sec + (size_t)b * HID + d, (es[r] / den[b]) * x[i]);
}
__global__ void pool_div_kernel(float* __restrict__ sec, const float* __restrict__ cnt, int n) {
  int i = blockIdx.x * 256 + threadIdx.x;
  if (i < n) sec[i] /= cnt[i >> 8];
}

// ------------------------------------------------ fusion concat
__global__ void concat_kernel(const float* __restrict__ x, const float* __restrict__ sec,
                              const int* __restrict__ seg, float* __restrict__ out, int n) {
  int i = blockIdx.x * 256 + threadIdx.x;
  if (i >= n) return;
  int r = i >> 9, d = i & 511;
  out[i] = (d < HID) ? x[(size_t)r * HID + d]
                     : sec[(size_t)seg[r] * HID + (d - HID)];
}

// =====================================================================
extern "C" void kernel_launch(void* const* d_in, const int* in_sizes, int n_in,
                              void* d_out, int out_size, void* d_ws, size_t ws_size,
                              hipStream_t stream) {
  auto F32 = [&](int i) { return (const float*)d_in[i]; };
  const float* x_stock  = F32(0);
  const int*   ei_stock = (const int*)d_in[1];
  const int*   batch    = (const int*)d_in[2];
  const int*   ei_sec   = (const int*)d_in[4];

  const int N  = in_sizes[2];
  const int E  = in_sizes[1] / 2;
  const int NS = in_sizes[3] / HID;
  const int ES = in_sizes[4] / 2;
  const int IN = in_sizes[0] / N;
  const int Etot = E + N, EStot = ES + NS;

  // param leaf indices (setup_inputs insertion order)
  enum {
    P_FN_G = 5, P_FN_B, P_IN_W, P_IN_B,
    P_SG0 = 9, P_SG1 = 17, P_XG0 = 25, P_XG1 = 33,   // Wl,bl,Wr,br,att,bias,ln_g,ln_b
    P_AP_W1 = 41, P_AP_B1, P_AP_W2, P_AP_B2,
    P_FU_W = 45, P_FU_B, P_FU_G, P_FU_B2,
    P_RH_W1 = 49, P_RH_B1, P_RH_W2, P_RH_B2, P_RH_W3, P_RH_B3,
    P_CH_W1 = 55, P_CH_B1, P_CH_W2, P_CH_B2,
    P_KH_W1 = 59, P_KH_B1, P_KH_W2, P_KH_B2,
    P_RG = 63, P_CL = 67, P_RK = 71
  };

  // ---------------- workspace carve-out
  size_t off = 0;
  char* base = (char*)d_ws;
  auto alloc = [&](size_t bytes) -> void* {
    off = (off + 255) & ~(size_t)255;
    void* p = base + off;
    off += bytes;
    return p;
  };
  auto allocH = [&](size_t elems) { return (_Float16*)alloc(elems * sizeof(_Float16)); };
  auto allocF = [&](size_t elems) { return (float*)alloc(elems * sizeof(float)); };
  auto allocU = [&](size_t elems) { return (unsigned*)alloc(elems * sizeof(unsigned)); };
  auto allocI = [&](size_t elems) { return (int*)alloc(elems * sizeof(int)); };

  _Float16* w_in16 = allocH((size_t)IN * HID);
  _Float16 *sg_wl16[2], *sg_wr16[2], *xg_wl16[2], *xg_wr16[2];
  for (int i = 0; i < 2; ++i) {
    sg_wl16[i] = allocH((size_t)HID * HID);
    sg_wr16[i] = allocH((size_t)HID * HID);
    xg_wl16[i] = allocH((size_t)HID * HID);
    xg_wr16[i] = allocH((size_t)HID * HID);
  }
  _Float16* ap1_16 = allocH((size_t)HID * 128);
  _Float16* fu16   = allocH((size_t)2 * HID * HID);
  _Float16* rh1_16 = allocH((size_t)HID * 128);
  _Float16* rh2_16 = allocH((size_t)128 * 64);
  _Float16* ch1_16 = allocH((size_t)HID * 128);
  _Float16* kh1_16 = allocH((size_t)HID * 128);
  _Float16* rg16   = allocH((size_t)HID * HID);
  _Float16* cl16   = allocH((size_t)HID * HID);
  _Float16* rk16   = allocH((size_t)HID * HID);

  float* bx = allocF((size_t)N * HID);        // stock features
  float* ba = allocF((size_t)N * HID);        // xl / rf / ...
  float* bb = allocF((size_t)N * HID);        // xr / cf / ...
  float* bc = allocF((size_t)N * 2 * HID);    // acc / concat / head temps
  float* be = allocF((size_t)Etot * 8);       // logits / ex
  unsigned* bmax = allocU((size_t)N * 8);
  float* bden = allocF((size_t)N * 8);
  float* bs = allocF(N);
  float* bw = allocF(N);
  int* esrc = allocI(Etot);
  int* edst = allocI(Etot);
  int* ssrc = allocI(EStot);
  int* sdst = allocI(EStot);
  float* bsec  = allocF((size_t)NS * HID);
  float* bsa   = allocF((size_t)NS * HID);
  float* bsb   = allocF((size_t)NS * HID);
  float* bsacc = allocF((size_t)NS * HID);
  float* bse   = allocF((size_t)EStot * 8);
  unsigned* bsmax = allocU((size_t)NS * 8);
  float* bsden = allocF((size_t)NS * 8);
  unsigned* pmax = allocU(NS);
  float* pden = allocF(NS);
  float* pcnt = allocF(NS);

  // ---------------- helpers
  auto fillf = [&](float* p, float v, size_t n) {
    fill_f32_kernel<<<dim3((unsigned)((n + 255) / 256)), 256, 0, stream>>>(p, v, n);
  };
  auto fillu = [&](unsigned* p, unsigned v, size_t n) {
    fill_u32_kernel<<<dim3((unsigned)((n + 255) / 256)), 256, 0, stream>>>(p, v, n);
  };
  auto cvt = [&](int pi, _Float16* o, int K, int Nout) {
    int n = K * Nout;
    cvt_transpose_kernel<<<dim3((n + 255) / 256), 256, 0, stream>>>(F32(pi), o, K, Nout);
  };
  auto GEMM = [&](int act, const float* A, const _Float16* Bt, const float* bias,
                  float* C, int M, int K, int Nout) {
    dim3 g((unsigned)((M + 127) / 128), (unsigned)(Nout / 64));
    if (act == 0)      wmma_gemm_kernel<0><<<g, 128, 0, stream>>>(A, Bt, bias, C, M, K, Nout);
    else if (act == 1) wmma_gemm_kernel<1><<<g, 128, 0, stream>>>(A, Bt, bias, C, M, K, Nout);
    else               wmma_gemm_kernel<2><<<g, 128, 0, stream>>>(A, Bt, bias, C, M, K, Nout);
  };
  auto LN = [&](const float* X, const float* res, const float* bvec, const float* g,
                const float* b, float* Y, int M, int D, int act) {
    ln_act_kernel<<<dim3((M + 3) / 4), 128, 0, stream>>>(X, res, bvec, g, b, Y, M, D, act);
  };
  auto HEAD = [&](const float* X, const float* W, const float* bias, float* o,
                  int M, int D, int Nout, int ostr, int mode) {
    small_head_kernel<<<dim3((M + 3) / 4), 128, 0, stream>>>(X, W, bias, o, M, D, Nout, ostr, mode);
  };
  auto gat_layer = [&](int pi, const _Float16* wl, const _Float16* wr, float* xbuf,
                       float* xl, float* xr, float* acc, float* elog, unsigned* mxe,
                       float* dne, const int* src, const int* dst, int nN, int nE) {
    GEMM(0, xbuf, wl, F32(pi + 1), xl, nN, HID, HID);
    GEMM(0, xbuf, wr, F32(pi + 3), xr, nN, HID, HID);
    fillu(mxe, ENC_NEG_INF, (size_t)nN * 8);
    fillf(dne, 0.f, (size_t)nN * 8);
    fillf(acc, 0.f, (size_t)nN * HID);
    edge_phase1_kernel<<<dim3((nE + 3) / 4), 128, 0, stream>>>(xl, xr, F32(pi + 4), src, dst, elog, mxe, nE);
    edge_phase2_kernel<<<dim3((nE * 8 + 255) / 256), 256, 0, stream>>>(elog, mxe, dne, dst, nE);
    edge_phase3_kernel<<<dim3((nE + 3) / 4), 128, 0, stream>>>(xl, elog, dne, src, dst, acc, nE);
    LN(acc, xbuf, F32(pi + 5), F32(pi + 6), F32(pi + 7), xbuf, nN, HID, 1);  // residual + ln + elu
  };

  // ---------------- weight prep (f32 [K][N] -> f16 [N][K])
  cvt(P_IN_W, w_in16, IN, HID);
  for (int i = 0; i < 2; ++i) {
    cvt(P_SG0 + 8 * i + 0, sg_wl16[i], HID, HID);
    cvt(P_SG0 + 8 * i + 2, sg_wr16[i], HID, HID);
    cvt(P_XG0 + 8 * i + 0, xg_wl16[i], HID, HID);
    cvt(P_XG0 + 8 * i + 2, xg_wr16[i], HID, HID);
  }
  cvt(P_AP_W1, ap1_16, HID, 128);
  cvt(P_FU_W, fu16, 2 * HID, HID);
  cvt(P_RH_W1, rh1_16, HID, 128);
  cvt(P_RH_W2, rh2_16, 128, 64);
  cvt(P_CH_W1, ch1_16, HID, 128);
  cvt(P_KH_W1, kh1_16, HID, 128);
  cvt(P_RG, rg16, HID, HID);
  cvt(P_CL, cl16, HID, HID);
  cvt(P_RK, rk16, HID, HID);

  // ---------------- edge lists with self loops
  build_edges_kernel<<<dim3((Etot + 255) / 256), 256, 0, stream>>>(ei_stock, esrc, edst, E, N);
  build_edges_kernel<<<dim3((EStot + 255) / 256), 256, 0, stream>>>(ei_sec, ssrc, sdst, ES, NS);

  // ---------------- input LN + projection
  LN(x_stock, nullptr, nullptr, F32(P_FN_G), F32(P_FN_B), bc, N, IN, 0);
  GEMM(0, bc, w_in16, F32(P_IN_B), bx, N, IN, HID);

  // ---------------- stock GAT x2
  gat_layer(P_SG0, sg_wl16[0], sg_wr16[0], bx, ba, bb, bc, be, bmax, bden, esrc, edst, N, Etot);
  gat_layer(P_SG1, sg_wl16[1], sg_wr16[1], bx, ba, bb, bc, be, bmax, bden, esrc, edst, N, Etot);

  // ---------------- attention pooling stock -> sector
  GEMM(2, bx, ap1_16, F32(P_AP_B1), bc, N, HID, 128);           // tanh(x@W1+b1)
  HEAD(bc, F32(P_AP_W2), F32(P_AP_B2), bs, N, 128, 1, 1, 0);    // scores s
  fillu(pmax, ENC_NEG_INF, NS);
  fillf(pden, 0.f, NS);
  fillf(pcnt, 0.f, NS);
  fillf(bsec, 0.f, (size_t)NS * HID);
  seg_max_kernel<<<dim3((N + 255) / 256), 256, 0, stream>>>(bs, batch, pmax, N);
  seg_exp_kernel<<<dim3((N + 255) / 256), 256, 0, stream>>>(bs, batch, pmax, bw, pden, pcnt, N);
  pool_acc_kernel<<<dim3((N * HID + 255) / 256), 256, 0, stream>>>(bx, bw, pden, batch, bsec, N * HID);
  pool_div_kernel<<<dim3((NS * HID + 255) / 256), 256, 0, stream>>>(bsec, pcnt, NS * HID);

  // ---------------- sector GAT x2
  gat_layer(P_XG0, xg_wl16[0], xg_wr16[0], bsec, bsa, bsb, bsacc, bse, bsmax, bsden, ssrc, sdst, NS, EStot);
  gat_layer(P_XG1, xg_wl16[1], xg_wr16[1], bsec, bsa, bsb, bsacc, bse, bsmax, bsden, ssrc, sdst, NS, EStot);

  // ---------------- fusion: comb = elu(ln(concat(x, sec[batch]) @ fu_W + fu_b))
  concat_kernel<<<dim3((N * 2 * HID + 255) / 256), 256, 0, stream>>>(bx, bsec, batch, bc, N * 2 * HID);
  GEMM(0, bc, fu16, F32(P_FU_B), ba, N, 2 * HID, HID);
  float* out_reg  = (float*)d_out;            // [N]
  float* out_clf  = out_reg + N;              // [N,2]
  float* out_rnk  = out_reg + 3 * (size_t)N;  // [N]
  float* out_comb = out_reg + 4 * (size_t)N;  // [N,256]
  LN(ba, nullptr, nullptr, F32(P_FU_G), F32(P_FU_B2), out_comb, N, HID, 1);

  // ---------------- regression head
  GEMM(0, out_comb, rg16, F32(P_RG + 1), bc, N, HID, HID);
  LN(bc, nullptr, nullptr, F32(P_RG + 2), F32(P_RG + 3), ba, N, HID, 2);   // rf = relu(ln(...))
  GEMM(1, ba, rh1_16, F32(P_RH_B1), bc, N, HID, 128);                      // h1
  float* bc2 = bc + (size_t)N * 128;
  GEMM(1, bc, rh2_16, F32(P_RH_B2), bc2, N, 128, 64);                      // h2
  HEAD(bc2, F32(P_RH_W3), F32(P_RH_B3), out_reg, N, 64, 1, 1, 1);          // tanh*0.1

  // ---------------- classification head
  GEMM(0, out_comb, cl16, F32(P_CL + 1), bc, N, HID, HID);
  LN(bc, nullptr, nullptr, F32(P_CL + 2), F32(P_CL + 3), bb, N, HID, 2);   // cf
  GEMM(1, bb, ch1_16, F32(P_CH_B1), bc, N, HID, 128);
  HEAD(bc, F32(P_CH_W2), F32(P_CH_B2), out_clf, N, 128, 2, 2, 0);

  // ---------------- ranking head
  GEMM(0, out_comb, rk16, F32(P_RK + 1), bc, N, HID, HID);
  LN(bc, nullptr, nullptr, F32(P_RK + 2), F32(P_RK + 3), bx, N, HID, 2);   // kf
  GEMM(1, bx, kh1_16, F32(P_KH_B1), bc, N, HID, 128);
  HEAD(bc, F32(P_KH_W2), F32(P_KH_B2), out_rnk, N, 128, 1, 1, 2);          // sigmoid

  (void)n_in; (void)out_size; (void)ws_size;
}
